// SlidingWindowAttention_60670708023693
// MI455X (gfx1250) — matrix operand order repaired and checked by reading, not compiled, not run on hardware
//
#include <hip/hip_runtime.h>
#include <hip/hip_bf16.h>

typedef __attribute__((ext_vector_type(16))) _Float16 v16h;
typedef __attribute__((ext_vector_type(8)))  float    v8f;
typedef __attribute__((ext_vector_type(4)))  float    f32x4;
typedef __attribute__((ext_vector_type(4)))  int      v4i;

#define BATCH  2
#define SEQ    2048
#define HID    2048
#define NHEADS 32
#define NKV    8
#define HDIM   64
#define WINDOW 512
#define KVDIM  (NKV * HDIM)     // 512
#define QDIM   (NHEADS * HDIM)  // 2048

#define WMMA_F16(a, b, c) \
  __builtin_amdgcn_wmma_f32_16x16x32_f16(false, (a), false, (b), (short)0, (c), false, false)

#if defined(__has_builtin)
#if __has_builtin(__builtin_amdgcn_global_load_async_to_lds_b128)
#define HAVE_ASYNC_LDS 1
#endif
#endif

// async-to-LDS builtin takes (v4i AS1*, v4i AS3*, imm offset, imm cpol)
#define GVEC(p) ((__attribute__((address_space(1))) v4i*)(p))
#define LVEC(p) ((__attribute__((address_space(3))) v4i*)(p))

// ---------------- fragment loaders (wave32 WMMA operands, contiguous k = 16*(lane/16) + s) ----

// A fragment: 16(M) x 32(K) f16, row-major source with leading dim `ld`.
// lanes 0-15: M=lane, K in {0..15}; lanes 16-31: M=lane-16, K in {16..31}
static __device__ inline v16h frag_a_f16(const _Float16* base, int ld) {
  const int lane = threadIdx.x & 31;
  const int row  = lane & 15;
  const int kb   = (lane >> 4) << 4;  // 0 or 16
  const _Float16* p = base + (size_t)row * ld + kb;
  v16h a;
#pragma unroll
  for (int i = 0; i < 8; ++i) a[i] = p[i];
#pragma unroll
  for (int i = 0; i < 8; ++i) a[8 + i] = p[8 + i];
  return a;
}

// B fragment: 32(K) x 16(N) f16 from a row-major [N, K] source (weight rows).
// lanes 0-15 hold K=0..15 of column N=lane; lanes 16-31 hold K=16..31.
static __device__ inline v16h frag_b_nk(const _Float16* base, int ld) {
  const int lane = threadIdx.x & 31;
  const int n  = lane & 15;
  const int kk = (lane >> 4) << 4;  // 0 or 16
  const _Float16* p = base + (size_t)n * ld + kk;
  v16h b;
#pragma unroll
  for (int i = 0; i < 16; ++i) b[i] = p[i];
  return b;
}

// B fragment for K^T in attention scores: column n = key (kb+n), K-dim = head dims.
static __device__ inline v16h frag_b_key(const _Float16* kbase, int kb, int dchunk) {
  const int lane = threadIdx.x & 31;
  const int n  = lane & 15;
  const int kk = (lane >> 4) << 4;
  int key = kb + n;
  if (key < 0) key = 0;
  if (key >= SEQ) key = SEQ - 1;  // clamped rows are masked to P=0 later
  const _Float16* p = kbase + (size_t)key * KVDIM + dchunk + kk;
  v16h b;
#pragma unroll
  for (int i = 0; i < 16; ++i) b[i] = p[i];
  return b;
}

// B fragment for V in P@V: K-dim = keys (rows of V), N = head-dim chunk.
static __device__ inline v16h frag_b_v(const _Float16* vbase, int kb, int dchunk) {
  const int lane = threadIdx.x & 31;
  const int n  = lane & 15;
  const int kk = (lane >> 4) << 4;
  v16h b;
#pragma unroll
  for (int i = 0; i < 16; ++i) {
    int key = kb + kk + i;
    if (key < 0) key = 0;
    if (key >= SEQ) key = SEQ - 1;  // P=0 for out-of-window keys
    b[i] = vbase[(size_t)key * KVDIM + dchunk + n];
  }
  return b;
}

// ---------------- elementwise kernels ----------------------------------------

__global__ void swa_f32_to_f16(const float* __restrict__ in, _Float16* __restrict__ out, long n) {
  long i = (long)blockIdx.x * blockDim.x + threadIdx.x;
  if (i < n) out[i] = (_Float16)in[i];
}

// Interleaved-pair rotary (reference: even/odd gather, concat halves) for Q and K.
__global__ void swa_rope(const float* __restrict__ qf, const float* __restrict__ kf,
                         const int* __restrict__ pos,
                         _Float16* __restrict__ qh, _Float16* __restrict__ kh) {
  long idx = (long)blockIdx.x * blockDim.x + threadIdx.x;
  const long total = (long)BATCH * SEQ * (NHEADS + NKV) * (HDIM / 2);
  if (idx >= total) return;
  const int t = (int)(idx & 31);            // rotary pair index 0..31
  long r = idx >> 5;
  const int head = (int)(r % (NHEADS + NKV));
  const long bs = r / (NHEADS + NKV);       // b*SEQ + s
  const float p = (float)pos[bs];
  // inv_freq[t] = 10000^(-2t/64) = exp(-t * ln(10000)/32)
  const float ang = p * __expf(-0.28782313662425572f * (float)t);
  const float c = __cosf(ang), s = __sinf(ang);
  if (head < NHEADS) {
    const float* src = qf + bs * QDIM + head * HDIM;
    _Float16* dst = qh + bs * QDIM + head * HDIM;
    const float e = src[2 * t], o = src[2 * t + 1];
    dst[t]      = (_Float16)(e * c - o * s);
    dst[32 + t] = (_Float16)(o * c + e * s);
  } else {
    const int kv = head - NHEADS;
    const float* src = kf + bs * KVDIM + kv * HDIM;
    _Float16* dst = kh + bs * KVDIM + kv * HDIM;
    const float e = src[2 * t], o = src[2 * t + 1];
    dst[t]      = (_Float16)(e * c - o * s);
    dst[32 + t] = (_Float16)(o * c + e * s);
  }
}

// ---------------- WMMA GEMM: C[M,N] (f32) = A[M,K] (f16) * W[N,K]^T (f16) -----
// Block tile 64x32, 8 waves (256 thr); wave w -> sub-tile rows (w>>1)*16, cols (w&1)*16
// LDS staging moves 128 bits per lane per issue (async-to-LDS when available).

#define GT_M 64
#define GT_N 32
#define GT_K 32
#define LDS_A 40
#define LDS_B 40

__global__ __launch_bounds__(256)
void swa_gemm_f16_f32(const _Float16* __restrict__ A, int lda,
                      const _Float16* __restrict__ Bw, int ldb,
                      float* __restrict__ C, int ldc, int K) {
  __shared__ _Float16 As[GT_M][LDS_A];
  __shared__ _Float16 Bs[GT_N][LDS_B];
  const int tid = threadIdx.x;
  const int wave = tid >> 5, lane = tid & 31;
  const int m0 = blockIdx.y * GT_M, n0 = blockIdx.x * GT_N;
  const int wm = (wave >> 1) * 16, wn = (wave & 1) * 16;

  // 16-byte staging chunks: A tile 64x32 halfs = 256 chunks; B tile 32x32 = 128.
  const int ar = tid >> 2;            // 0..63
  const int ac = (tid & 3) * 8;       // 0,8,16,24  (byte offset 2*ac is 16B aligned)

  v8f acc = {};
  for (int k0 = 0; k0 < K; k0 += GT_K) {
    const _Float16* asrc = A + (size_t)(m0 + ar) * lda + k0 + ac;
#if defined(HAVE_ASYNC_LDS)
    __builtin_amdgcn_global_load_async_to_lds_b128(GVEC(asrc), LVEC(&As[ar][ac]), 0, 0);
    if (tid < 128) {
      const _Float16* bsrc = Bw + (size_t)(n0 + (tid >> 2)) * ldb + k0 + ac;
      __builtin_amdgcn_global_load_async_to_lds_b128(GVEC(bsrc), LVEC(&Bs[tid >> 2][ac]), 0, 0);
    }
    asm volatile("s_wait_asynccnt 0" ::: "memory");
#else
    *(f32x4*)&As[ar][ac] = *(const f32x4*)asrc;
    if (tid < 128) {
      const _Float16* bsrc = Bw + (size_t)(n0 + (tid >> 2)) * ldb + k0 + ac;
      *(f32x4*)&Bs[tid >> 2][ac] = *(const f32x4*)bsrc;
    }
#endif
    __builtin_prefetch(&A[(size_t)(m0 + ar) * lda + k0 + GT_K], 0, 1);
    __syncthreads();
    v16h a = frag_a_f16(&As[wm][0], LDS_A);
    v16h b = frag_b_nk(&Bs[wn][0], LDS_B);
    acc = WMMA_F16(a, b, acc);
    __syncthreads();
  }
  const int half = lane >> 4, col = lane & 15;
#pragma unroll
  for (int r = 0; r < 8; ++r) {
    int row = wm + r + 8 * half;
    C[(size_t)(m0 + row) * ldc + n0 + wn + col] = acc[r];
  }
}

// ---------------- flash sliding-window attention ------------------------------
// One wave per 16-query tile of one (b, head). 4 independent waves per block;
// no block barriers inside the loop (trip counts differ) — wave-local DS waits.

#define AW 4

__global__ __launch_bounds__(AW * 32)
void swa_attn(const _Float16* __restrict__ q, const _Float16* __restrict__ k,
              const _Float16* __restrict__ v, _Float16* __restrict__ out) {
  __shared__ float    Stile[AW][16][33];
  __shared__ _Float16 Ptile[AW][16][40];
  __shared__ float    Mrow[AW][16], Lrow[AW][16], Arow[AW][16];

  const int w = threadIdx.x >> 5;
  const int lane = threadIdx.x & 31;
  const int wid = blockIdx.x * AW + w;
  const int QT = SEQ / 16;
  const int qt = wid % QT;
  const int h  = (wid / QT) % NHEADS;
  const int b  = wid / (QT * NHEADS);
  const int i0 = qt * 16;
  const int kvh = h >> 2;  // 4 query heads per KV head

  const _Float16* qbase = q + ((size_t)(b * SEQ + i0)) * QDIM + h * HDIM;
  const _Float16* kbase = k + ((size_t)b * SEQ) * KVDIM + kvh * HDIM;
  const _Float16* vbase = v + ((size_t)b * SEQ) * KVDIM + kvh * HDIM;

  const v16h aq0 = frag_a_f16(qbase, QDIM);       // d = 0..31
  const v16h aq1 = frag_a_f16(qbase + 32, QDIM);  // d = 32..63

  if (lane < 16) { Mrow[w][lane] = -3.0e38f; Lrow[w][lane] = 0.0f; }
  v8f o0 = {}, o1 = {}, o2 = {}, o3 = {};

  const int half = lane >> 4, col = lane & 15;
  const int kend = i0 + 15;
  int kbeg = i0 - WINDOW + 1; if (kbeg < 0) kbeg = 0;

  for (int kb = kbeg & ~31; kb <= kend; kb += 32) {
    // scores: two 16x16 key tiles, K-dim 64 split in two 32-chunks
    v16h bk00 = frag_b_key(kbase, kb,       0);
    v16h bk01 = frag_b_key(kbase, kb,      32);
    v16h bk10 = frag_b_key(kbase, kb + 16,  0);
    v16h bk11 = frag_b_key(kbase, kb + 16, 32);
    v8f z = {};
    v8f s0 = WMMA_F16(aq0, bk00, z); s0 = WMMA_F16(aq1, bk01, s0);
    v8f s1 = WMMA_F16(aq0, bk10, z); s1 = WMMA_F16(aq1, bk11, s1);

    // scale + window mask, stage to LDS
#pragma unroll
    for (int r = 0; r < 8; ++r) {
      const int row = r + 8 * half;
      const int i = i0 + row;
      const int j0 = kb + col, j1 = kb + 16 + col;
      Stile[w][row][col]      = ((j0 <= i) && (j0 >= i - WINDOW + 1)) ? s0[r] * 0.125f : -1.0e30f;
      Stile[w][row][col + 16] = ((j1 <= i) && (j1 >= i - WINDOW + 1)) ? s1[r] * 0.125f : -1.0e30f;
    }
    asm volatile("s_wait_dscnt 0" ::: "memory");

    // online softmax: one lane per query row
    if (lane < 16) {
      const int row = lane;
      const float m_old = Mrow[w][row];
      float mx = m_old;
#pragma unroll
      for (int c = 0; c < 32; ++c) mx = fmaxf(mx, Stile[w][row][c]);
      const float alpha = __expf(m_old - mx);
      float sum = 0.0f;
#pragma unroll
      for (int c = 0; c < 32; ++c) {
        const float p = __expf(Stile[w][row][c] - mx);
        Ptile[w][row][c] = (_Float16)p;
        sum += p;
      }
      Mrow[w][row] = mx;
      Lrow[w][row] = Lrow[w][row] * alpha + sum;
      Arow[w][row] = alpha;
    }
    asm volatile("s_wait_dscnt 0" ::: "memory");

    float al[8];
#pragma unroll
    for (int r = 0; r < 8; ++r) al[r] = Arow[w][r + 8 * half];
#pragma unroll
    for (int r = 0; r < 8; ++r) {
      o0[r] *= al[r]; o1[r] *= al[r]; o2[r] *= al[r]; o3[r] *= al[r];
    }

    // P (16x32) @ V (32x64): 4 WMMAs over head-dim chunks
    const v16h pf = frag_a_f16(&Ptile[w][0][0], 40);
    o0 = WMMA_F16(pf, frag_b_v(vbase, kb,  0), o0);
    o1 = WMMA_F16(pf, frag_b_v(vbase, kb, 16), o1);
    o2 = WMMA_F16(pf, frag_b_v(vbase, kb, 32), o2);
    o3 = WMMA_F16(pf, frag_b_v(vbase, kb, 48), o3);
  }

  asm volatile("s_wait_dscnt 0" ::: "memory");
  float il[8];
#pragma unroll
  for (int r = 0; r < 8; ++r) il[r] = 1.0f / Lrow[w][r + 8 * half];

  _Float16* obase = out + ((size_t)(b * SEQ + i0)) * QDIM + h * HDIM;
#pragma unroll
  for (int r = 0; r < 8; ++r) {
    const int row = r + 8 * half;
    obase[(size_t)row * QDIM +  0 + col] = (_Float16)(o0[r] * il[r]);
    obase[(size_t)row * QDIM + 16 + col] = (_Float16)(o1[r] * il[r]);
    obase[(size_t)row * QDIM + 32 + col] = (_Float16)(o2[r] * il[r]);
    obase[(size_t)row * QDIM + 48 + col] = (_Float16)(o3[r] * il[r]);
  }
}

// ---------------- launch ------------------------------------------------------

extern "C" void kernel_launch(void* const* d_in, const int* in_sizes, int n_in,
                              void* d_out, int out_size, void* d_ws, size_t ws_size,
                              hipStream_t stream) {
  (void)in_sizes; (void)n_in; (void)out_size; (void)ws_size;
  const float* X  = (const float*)d_in[0];
  const float* Wq = (const float*)d_in[1];
  const float* Wk = (const float*)d_in[2];
  const float* Wv = (const float*)d_in[3];
  const float* Wo = (const float*)d_in[4];
  const int*  pos = (const int*)d_in[5];
  float* out = (float*)d_out;

  char* ws = (char*)d_ws;
  size_t off = 0;
  auto alloc = [&](size_t bytes) -> void* {
    off = (off + 255) & ~(size_t)255;
    void* p = ws + off;
    off += bytes;
    return p;
  };
  const size_t BS = (size_t)BATCH * SEQ;  // 4096

  _Float16* xh  = (_Float16*)alloc(BS * HID * 2);
  _Float16* wqh = (_Float16*)alloc((size_t)QDIM * HID * 2);
  _Float16* wkh = (_Float16*)alloc((size_t)KVDIM * HID * 2);
  _Float16* wvh = (_Float16*)alloc((size_t)KVDIM * HID * 2);
  _Float16* woh = (_Float16*)alloc((size_t)HID * QDIM * 2);
  float*    qf  = (float*)alloc(BS * QDIM * 4);
  float*    kf  = (float*)alloc(BS * KVDIM * 4);
  float*    vf  = (float*)alloc(BS * KVDIM * 4);
  _Float16* qh  = (_Float16*)alloc(BS * QDIM * 2);
  _Float16* kh  = (_Float16*)alloc(BS * KVDIM * 2);
  _Float16* vh  = (_Float16*)alloc(BS * KVDIM * 2);
  _Float16* ah  = (_Float16*)alloc(BS * QDIM * 2);

  auto cvt = [&](const float* src, _Float16* dst, long n) {
    swa_f32_to_f16<<<(unsigned)((n + 255) / 256), 256, 0, stream>>>(src, dst, n);
  };
  cvt(X,  xh,  (long)BS * HID);
  cvt(Wq, wqh, (long)QDIM * HID);
  cvt(Wk, wkh, (long)KVDIM * HID);
  cvt(Wv, wvh, (long)KVDIM * HID);
  cvt(Wo, woh, (long)HID * QDIM);

  // projections: Q/K/V = X @ W^T
  swa_gemm_f16_f32<<<dim3(QDIM / GT_N,  (unsigned)(BS / GT_M)), 256, 0, stream>>>(
      xh, HID, wqh, HID, qf, QDIM, HID);
  swa_gemm_f16_f32<<<dim3(KVDIM / GT_N, (unsigned)(BS / GT_M)), 256, 0, stream>>>(
      xh, HID, wkh, HID, kf, KVDIM, HID);
  swa_gemm_f16_f32<<<dim3(KVDIM / GT_N, (unsigned)(BS / GT_M)), 256, 0, stream>>>(
      xh, HID, wvh, HID, vf, KVDIM, HID);

  const long nrope = (long)BATCH * SEQ * (NHEADS + NKV) * (HDIM / 2);
  swa_rope<<<(unsigned)((nrope + 255) / 256), 256, 0, stream>>>(qf, kf, pos, qh, kh);
  cvt(vf, vh, (long)BS * KVDIM);

  const int nwaves = BATCH * NHEADS * (SEQ / 16);  // 8192
  swa_attn<<<nwaves / AW, AW * 32, 0, stream>>>(qh, kh, vh, ah);

  // output projection: out = attn @ Wo^T
  swa_gemm_f16_f32<<<dim3(HID / GT_N, (unsigned)(BS / GT_M)), 256, 0, stream>>>(
      ah, QDIM, woh, QDIM, out, HID, QDIM);
}
